// Conv_29411936043447
// MI455X (gfx1250) — matrix operand order repaired and checked by reading, not compiled, not run on hardware
//
#include <hip/hip_runtime.h>

typedef __attribute__((ext_vector_type(16))) _Float16 v16h;
typedef __attribute__((ext_vector_type(8)))  _Float16 v8h;
typedef __attribute__((ext_vector_type(8)))  float    v8f;

#define T_DIM    4
#define N_CELLS  49152
#define K_NB     9
#define F_IN     64
#define F_OUT    64
#define K_TOT    (K_NB * F_IN)               // 576
#define K_STEPS  (K_TOT / 32)                // 18 WMMA K-steps of 32
#define WFRAG_ELEMS (K_STEPS * 4 * 32 * 16)  // 36864 f16 = 73728 B
#define NROWS    (T_DIM * N_CELLS)           // 196608
#define XH_ELEMS (NROWS * F_IN)              // 12582912 f16
#define WS_NEEDED ((size_t)WFRAG_ELEMS * 2 + (size_t)XH_ELEMS * 2)

// ---------------------------------------------------------------------------
// Kernel 1: repack W [9,64,64] f32 -> f16 in WMMA-B fragment-ready layout.
// wfrag[((s*4 + c)*32 + lane)*16 + i]
//   = W_flat[ s*32 + (lane>=16 ? 8:0) + (i&7) + (i>=8 ? 16:0) , c*16 + (lane&15) ]
// so each lane's B fragment is 32 contiguous bytes (two ds_load_b128).
// ---------------------------------------------------------------------------
__global__ void repack_w_kernel(const float* __restrict__ W,
                                _Float16* __restrict__ wfrag) {
    int e = blockIdx.x * blockDim.x + threadIdx.x;
    if (e >= WFRAG_ELEMS) return;
    int i    = e & 15;
    int lane = (e >> 4) & 31;
    int c    = (e >> 9) & 3;
    int s    = e >> 11;
    int k    = s * 32 + ((lane >= 16) ? 8 : 0) + (i & 7) + ((i >= 8) ? 16 : 0);   // K index in [0,576)
    int o    = c * 16 + (lane & 15);                   // output column
    int knb  = k >> 6;
    int f    = k & 63;
    wfrag[e] = (_Float16)W[(knb * F_IN + f) * F_OUT + o];
}

// ---------------------------------------------------------------------------
// Kernel 2: stream-convert x f32 -> f16 (halves all gathered A traffic).
// ---------------------------------------------------------------------------
__global__ void convert_x_kernel(const float* __restrict__ x,
                                 _Float16* __restrict__ xh) {
    int e = (blockIdx.x * blockDim.x + threadIdx.x) * 8;
    if (e >= XH_ELEMS) return;
    float4 a = *(const float4*)(x + e);
    float4 b = *(const float4*)(x + e + 4);
    v8h h;
    h[0] = (_Float16)a.x; h[1] = (_Float16)a.y;
    h[2] = (_Float16)a.z; h[3] = (_Float16)a.w;
    h[4] = (_Float16)b.x; h[5] = (_Float16)b.y;
    h[6] = (_Float16)b.z; h[7] = (_Float16)b.w;
    *(v8h*)(xh + e) = h;
}

__device__ __forceinline__ v16h cat8(v8h lo, v8h hi) {
    return __builtin_shufflevector(lo, hi, 0, 1, 2, 3, 4, 5, 6, 7,
                                           8, 9, 10, 11, 12, 13, 14, 15);
}

__device__ __forceinline__ v8h cvt8(float4 a, float4 b) {
    v8h h;
    h[0] = (_Float16)a.x; h[1] = (_Float16)a.y;
    h[2] = (_Float16)a.z; h[3] = (_Float16)a.w;
    h[4] = (_Float16)b.x; h[5] = (_Float16)b.y;
    h[6] = (_Float16)b.z; h[7] = (_Float16)b.w;
    return h;
}

// ---------------------------------------------------------------------------
// Kernel 3: gathered GEMM. One wave = TWO 16-row tiles x 64 cols.
// Fully unrolled over the 9 neighbors: each neighbor iteration computes ONE
// base address per tile, issues 4 global_load_b128 at immediate offsets
// {0,32,64,96}B covering two K-steps, then 16 WMMAs. No indexed register
// arrays -> no v_movrels / hazard NOPs. Bias folded into accumulator init.
// ---------------------------------------------------------------------------
template <bool XF16>
__global__ void __launch_bounds__(256)
conv_gather_wmma_kernel(const float* __restrict__ x,
                        const _Float16* __restrict__ xh,
                        const int*   __restrict__ adjc,
                        const _Float16* __restrict__ wfrag,
                        const float* __restrict__ bias,
                        float* __restrict__ out) {
    extern __shared__ __align__(16) unsigned char smem_raw[];
    _Float16* w_lds = (_Float16*)smem_raw;

    // Cooperative copy of pre-swizzled W into LDS (73728 B = 4608 float4)
    {
        const float4* src = (const float4*)wfrag;
        float4*       dst = (float4*)w_lds;
        for (int i = threadIdx.x; i < (WFRAG_ELEMS * 2) / 16; i += 256)
            dst[i] = src[i];
    }
    __syncthreads();

    const int lane   = threadIdx.x & 31;
    const int wave   = threadIdx.x >> 5;
    const int lanelo = lane & 15;
    const int hi     = (lane < 16) ? 0 : 8;   // A chunk offset / D row offset

    // 32-row super-tile per wave over flattened (t,n); 32 | N so same t.
    const int grow0 = blockIdx.x * 256 + wave * 32;
    const int t     = grow0 / N_CELLS;
    const int n0    = (grow0 % N_CELLS) + lanelo;       // tile 0 row
    const int n1    = n0 + 16;                          // tile 1 row

    int nbr0[K_NB], nbr1[K_NB];
#pragma unroll
    for (int k = 0; k < K_NB; ++k) nbr0[k] = adjc[n0 * K_NB + k];
#pragma unroll
    for (int k = 0; k < K_NB; ++k) nbr1[k] = adjc[n1 * K_NB + k];

    // Bias folded into accumulators: C col N = lane&15 for every C VGPR.
    v8f acc[2][4];
#pragma unroll
    for (int c = 0; c < 4; ++c) {
        float bv = bias[c * 16 + lanelo];
        acc[0][c] = (v8f)(bv);
        acc[1][c] = (v8f)(bv);
    }

    const unsigned xT = (unsigned)t * N_CELLS;   // 32-bit indices throughout

#pragma unroll
    for (int knb = 0; knb < K_NB; ++knb) {
        // One base address per tile; 4 chunks at immediate offsets feed
        // BOTH K-steps of this neighbor (features hi+{0..7,16..23,32..39,48..55}).
        v16h a0[2], a1[2];
        if (XF16) {
            const unsigned i0 = ((xT + (unsigned)nbr0[knb]) << 6) + hi;
            const unsigned i1 = ((xT + (unsigned)nbr1[knb]) << 6) + hi;
            v8h q00 = *(const v8h*)(xh + i0);
            v8h q01 = *(const v8h*)(xh + i0 + 16);
            v8h q02 = *(const v8h*)(xh + i0 + 32);
            v8h q03 = *(const v8h*)(xh + i0 + 48);
            v8h q10 = *(const v8h*)(xh + i1);
            v8h q11 = *(const v8h*)(xh + i1 + 16);
            v8h q12 = *(const v8h*)(xh + i1 + 32);
            v8h q13 = *(const v8h*)(xh + i1 + 48);
            a0[0] = cat8(q00, q01); a0[1] = cat8(q02, q03);
            a1[0] = cat8(q10, q11); a1[1] = cat8(q12, q13);
        } else {
            const unsigned i0 = ((xT + (unsigned)nbr0[knb]) << 6) + hi;
            const unsigned i1 = ((xT + (unsigned)nbr1[knb]) << 6) + hi;
            const float* p0 = x + i0;
            const float* p1 = x + i1;
            a0[0] = cat8(cvt8(*(const float4*)(p0 +  0), *(const float4*)(p0 +  4)),
                         cvt8(*(const float4*)(p0 + 16), *(const float4*)(p0 + 20)));
            a0[1] = cat8(cvt8(*(const float4*)(p0 + 32), *(const float4*)(p0 + 36)),
                         cvt8(*(const float4*)(p0 + 48), *(const float4*)(p0 + 52)));
            a1[0] = cat8(cvt8(*(const float4*)(p1 +  0), *(const float4*)(p1 +  4)),
                         cvt8(*(const float4*)(p1 + 16), *(const float4*)(p1 + 20)));
            a1[1] = cat8(cvt8(*(const float4*)(p1 + 32), *(const float4*)(p1 + 36)),
                         cvt8(*(const float4*)(p1 + 48), *(const float4*)(p1 + 52)));
        }

#pragma unroll
        for (int h2 = 0; h2 < 2; ++h2) {           // the two K-steps (f0 = 0, 32)
            const int s = knb * 2 + h2;
            const _Float16* wbase = w_lds + ((s * 4) * 32 + lane) * 16;
#pragma unroll
            for (int c = 0; c < 4; ++c) {
                v16h b = *(const v16h*)(wbase + c * 32 * 16);
                acc[0][c] = __builtin_amdgcn_wmma_f32_16x16x32_f16(
                    false, a0[h2], false, b, (short)0, acc[0][c], false, false);
                acc[1][c] = __builtin_amdgcn_wmma_f32_16x16x32_f16(
                    false, a1[h2], false, b, (short)0, acc[1][c], false, false);
            }
        }
    }

    // Epilogue: C/D 16x16 f32 layout: VGPR r -> M = r + hi, N = lane&15.
    const unsigned ob0 = (unsigned)grow0 * F_OUT;
    const unsigned ob1 = (unsigned)(grow0 + 16) * F_OUT;
#pragma unroll
    for (int c = 0; c < 4; ++c) {
        const int ocol = c * 16 + lanelo;
#pragma unroll
        for (int r = 0; r < 8; ++r) {
            out[ob0 + (unsigned)(hi + r) * F_OUT + ocol] = acc[0][c][r];
            out[ob1 + (unsigned)(hi + r) * F_OUT + ocol] = acc[1][c][r];
        }
    }
}

// ---------------------------------------------------------------------------
extern "C" void kernel_launch(void* const* d_in, const int* in_sizes, int n_in,
                              void* d_out, int out_size, void* d_ws, size_t ws_size,
                              hipStream_t stream) {
    const float* x    = (const float*)d_in[0];   // [1,1,4,49152,64] f32
    const int*   adjc = (const int*)  d_in[1];   // [49152,9] i32
    const float* W    = (const float*)d_in[2];   // [9,64,64] f32
    const float* b    = (const float*)d_in[3];   // [64] f32
    float*       out  = (float*)d_out;           // [1,1,4,49152,64] f32

    _Float16* wfrag = (_Float16*)d_ws;                      // 73728 B
    _Float16* xh    = (_Float16*)((char*)d_ws + (size_t)WFRAG_ELEMS * 2);

    const bool use_f16x = (ws_size >= WS_NEEDED);           // fixed per run

    repack_w_kernel<<<WFRAG_ELEMS / 256, 256, 0, stream>>>(W, wfrag);

    const size_t lds_bytes = (size_t)WFRAG_ELEMS * sizeof(_Float16);  // 73728
    const int nblocks = NROWS / 256;   // 768 blocks, 8 waves x 32 rows each

    if (use_f16x) {
        convert_x_kernel<<<XH_ELEMS / (256 * 8), 256, 0, stream>>>(x, xh);
        (void)hipFuncSetAttribute(
            reinterpret_cast<const void*>(&conv_gather_wmma_kernel<true>),
            hipFuncAttributeMaxDynamicSharedMemorySize, (int)lds_bytes);
        conv_gather_wmma_kernel<true><<<nblocks, 256, lds_bytes, stream>>>(
            x, xh, adjc, wfrag, b, out);
    } else {
        (void)hipFuncSetAttribute(
            reinterpret_cast<const void*>(&conv_gather_wmma_kernel<false>),
            hipFuncAttributeMaxDynamicSharedMemorySize, (int)lds_bytes);
        conv_gather_wmma_kernel<false><<<nblocks, 256, lds_bytes, stream>>>(
            x, nullptr, adjc, wfrag, b, out);
    }
}
